// CrossEntropy_54322746360602
// MI455X (gfx1250) — compile-verified, matrix operand order checked by reading, not run on hardware
//
#include <hip/hip_runtime.h>
#include <hip/hip_bf16.h>
#include <math.h>

#define BN   8192      // batch (rows of A, rows of B, and both dims of cos)
#define DK   1024      // feature dim
#define EPSV 1e-8f

typedef __attribute__((ext_vector_type(16))) __bf16        v16bf;
typedef __attribute__((ext_vector_type(8)))  float         v8f;
typedef __attribute__((ext_vector_type(4)))  unsigned int  u32x4;

union FragBF {
    v16bf bf;
    u32x4 q[2];
};

// ---------------- helpers: fp32 -> bf16 (RNE) split ----------------
__device__ __forceinline__ unsigned int f2bf_bits(float f) {
    unsigned int u = __float_as_uint(f);
    u += 0x7FFFu + ((u >> 16) & 1u);
    return u >> 16;
}
__device__ __forceinline__ float bf2f(unsigned int h) {
    return __uint_as_float(h << 16);
}

// ============ kernel 1: one-time split-bf16 decomposition + row norms ============
// Each block handles one row (1024 floats) of A or B: writes bf16 hi/lo planes
// and the row L2 norm. This removes ALL conversion VALU from the GEMM hot loop
// (each element would otherwise be re-converted by 64 tile blocks).
__global__ __launch_bounds__(256)
void preprocess_kernel(const float* __restrict__ A, const float* __restrict__ Bf,
                       unsigned short* __restrict__ Ahi, unsigned short* __restrict__ Alo,
                       unsigned short* __restrict__ Bhi, unsigned short* __restrict__ Blo,
                       float* __restrict__ na, float* __restrict__ nb) {
    __shared__ float red[256];
    const int r   = blockIdx.x;
    const int tid = threadIdx.x;

    const float* X;
    unsigned short *Hp, *Lp;
    float* dst;
    int row;
    if (r < BN) {
        row = r;
        X  = A   + (size_t)row * DK;
        Hp = Ahi + (size_t)row * DK;
        Lp = Alo + (size_t)row * DK;
        dst = na;
    } else {
        row = r - BN;
        X  = Bf  + (size_t)row * DK;
        Hp = Bhi + (size_t)row * DK;
        Lp = Blo + (size_t)row * DK;
        dst = nb;
    }

    const int c = tid * 4;                       // 256 threads x 4 = 1024
    float4 v = *reinterpret_cast<const float4*>(X + c);

    float s = v.x * v.x + v.y * v.y + v.z * v.z + v.w * v.w;

    unsigned int h0 = f2bf_bits(v.x), h1 = f2bf_bits(v.y);
    unsigned int h2 = f2bf_bits(v.z), h3 = f2bf_bits(v.w);
    unsigned int l0 = f2bf_bits(v.x - bf2f(h0));
    unsigned int l1 = f2bf_bits(v.y - bf2f(h1));
    unsigned int l2 = f2bf_bits(v.z - bf2f(h2));
    unsigned int l3 = f2bf_bits(v.w - bf2f(h3));
    uint2 ph; ph.x = h0 | (h1 << 16); ph.y = h2 | (h3 << 16);
    uint2 pl; pl.x = l0 | (l1 << 16); pl.y = l2 | (l3 << 16);
    *reinterpret_cast<uint2*>(Hp + c) = ph;
    *reinterpret_cast<uint2*>(Lp + c) = pl;

    red[tid] = s;
    __syncthreads();
    for (int off = 128; off > 0; off >>= 1) {
        if (tid < off) red[tid] += red[tid + off];
        __syncthreads();
    }
    if (tid == 0) dst[row] = sqrtf(red[0]);
}

// ============ kernel 2: cos = (A @ B^T) / max(na*nb, eps) ============
// 128x128 C-tile per 256-thread (8-wave) block, split-bf16 (hi*hi + hi*lo + lo*hi).
// Hot loop is pure data movement + WMMA: b128 global loads of pre-split bf16,
// b128 LDS stores, b128 fragment loads, 24 v_wmma per chunk. Double-buffered LDS
// with an explicitly 2x-unrolled pipeline so buffer offsets are immediates.
#define LDS_STRIDE 40   // 32 bf16 + 8 pad (16B-aligned rows, no bank conflicts)
#define TILE_ELEMS (128 * LDS_STRIDE)

__global__ __launch_bounds__(256)
void gemm_cos_kernel(const unsigned short* __restrict__ Ahi, const unsigned short* __restrict__ Alo,
                     const unsigned short* __restrict__ Bhi, const unsigned short* __restrict__ Blo,
                     const float* __restrict__ na, const float* __restrict__ nb,
                     float* __restrict__ cosOut) {
    // [buffer][hi=0 / lo=1]
    __shared__ unsigned short sA[2][2][TILE_ELEMS];
    __shared__ unsigned short sB[2][2][TILE_ELEMS];

    const int tid     = threadIdx.x;
    const int lane    = tid & 31;          // wave32
    const int wave    = tid >> 5;          // 8 waves
    const int waveM   = wave & 3;          // 4 waves along M
    const int waveN   = wave >> 2;         // 2 waves along N
    const int laneRow = lane & 15;
    const int kgrp    = lane >> 4;         // 0 or 1 (K half select)

    const int tileM = blockIdx.y * 128;
    const int tileN = blockIdx.x * 128;

    v8f acc[2][4] = {};   // 2 (M subtiles) x 4 (N subtiles) of 16x16 f32

    const unsigned short* gAhi = Ahi + (size_t)tileM * DK;
    const unsigned short* gAlo = Alo + (size_t)tileM * DK;
    const unsigned short* gBhi = Bhi + (size_t)tileN * DK;
    const unsigned short* gBlo = Blo + (size_t)tileN * DK;

    // staging registers: 8 x b128 in flight (software pipeline)
    u32x4 rAh[2], rAl[2], rBh[2], rBl[2];

    auto load_regs = [&](int k0) {
#pragma unroll
        for (int i = 0; i < 2; ++i) {
            int idx = tid + i * 256;            // 0..511 (128 rows x 4 b128-segs)
            int row = idx >> 2;                 // 0..127
            int sgo = (idx & 3) * 8;            // bf16 offset 0,8,16,24
            size_t g = (size_t)row * DK + k0 + sgo;
            rAh[i] = *reinterpret_cast<const u32x4*>(gAhi + g);
            rAl[i] = *reinterpret_cast<const u32x4*>(gAlo + g);
            rBh[i] = *reinterpret_cast<const u32x4*>(gBhi + g);
            rBl[i] = *reinterpret_cast<const u32x4*>(gBlo + g);
        }
    };
    auto store_lds = [&](int buf) {
#pragma unroll
        for (int i = 0; i < 2; ++i) {
            int idx = tid + i * 256;
            int row = idx >> 2;
            int sgo = (idx & 3) * 8;
            int off = row * LDS_STRIDE + sgo;
            *reinterpret_cast<u32x4*>(&sA[buf][0][off]) = rAh[i];
            *reinterpret_cast<u32x4*>(&sA[buf][1][off]) = rAl[i];
            *reinterpret_cast<u32x4*>(&sB[buf][0][off]) = rBh[i];
            *reinterpret_cast<u32x4*>(&sB[buf][1][off]) = rBl[i];
        }
    };

    // fragment LDS offsets (bf16 units), buffer-invariant
    int offA[2], offB[4];
#pragma unroll
    for (int mi = 0; mi < 2; ++mi)
        offA[mi] = (waveM * 32 + mi * 16 + laneRow) * LDS_STRIDE + kgrp * 8;
#pragma unroll
    for (int ni = 0; ni < 4; ++ni)
        offB[ni] = (waveN * 64 + ni * 16 + laneRow) * LDS_STRIDE + kgrp * 8;

    auto compute = [&](int buf) {
        const unsigned short* pAhi = sA[buf][0];
        const unsigned short* pAlo = sA[buf][1];
        const unsigned short* pBhi = sB[buf][0];
        const unsigned short* pBlo = sB[buf][1];
        FragBF aHi[2], aLo[2], bHi[4], bLo[4];
#pragma unroll
        for (int mi = 0; mi < 2; ++mi) {
            aHi[mi].q[0] = *reinterpret_cast<const u32x4*>(&pAhi[offA[mi]]);
            aHi[mi].q[1] = *reinterpret_cast<const u32x4*>(&pAhi[offA[mi] + 16]);
            aLo[mi].q[0] = *reinterpret_cast<const u32x4*>(&pAlo[offA[mi]]);
            aLo[mi].q[1] = *reinterpret_cast<const u32x4*>(&pAlo[offA[mi] + 16]);
        }
#pragma unroll
        for (int ni = 0; ni < 4; ++ni) {
            bHi[ni].q[0] = *reinterpret_cast<const u32x4*>(&pBhi[offB[ni]]);
            bHi[ni].q[1] = *reinterpret_cast<const u32x4*>(&pBhi[offB[ni] + 16]);
            bLo[ni].q[0] = *reinterpret_cast<const u32x4*>(&pBlo[offB[ni]]);
            bLo[ni].q[1] = *reinterpret_cast<const u32x4*>(&pBlo[offB[ni] + 16]);
        }
#pragma unroll
        for (int mi = 0; mi < 2; ++mi) {
#pragma unroll
            for (int ni = 0; ni < 4; ++ni) {
                acc[mi][ni] = __builtin_amdgcn_wmma_f32_16x16x32_bf16(
                    false, aHi[mi].bf, false, bHi[ni].bf, (short)0, acc[mi][ni], false, false);
                acc[mi][ni] = __builtin_amdgcn_wmma_f32_16x16x32_bf16(
                    false, aHi[mi].bf, false, bLo[ni].bf, (short)0, acc[mi][ni], false, false);
                acc[mi][ni] = __builtin_amdgcn_wmma_f32_16x16x32_bf16(
                    false, aLo[mi].bf, false, bHi[ni].bf, (short)0, acc[mi][ni], false, false);
            }
        }
    };

    const int NCHUNK = DK / 32;   // 32, even
    // prologue: fill buffer 0 with chunk 0
    load_regs(0);
    store_lds(0);
    __syncthreads();

    for (int kc = 0; kc < NCHUNK; kc += 2) {
        // even chunk: compute from buf0, prefetch chunk kc+1 into buf1
        if (kc + 1 < NCHUNK) load_regs((kc + 1) * 32);
        compute(0);
        if (kc + 1 < NCHUNK) store_lds(1);
        __syncthreads();
        // odd chunk: compute from buf1, prefetch chunk kc+2 into buf0
        if (kc + 1 < NCHUNK) {
            if (kc + 2 < NCHUNK) load_regs((kc + 2) * 32);
            compute(1);
            if (kc + 2 < NCHUNK) store_lds(0);
        }
        __syncthreads();
    }

    // ---- epilogue: divide by max(||a||*||b||, eps), store cos ----
    // C layout: VGPR r, lane l -> row = r + 8*(l>>4), col = l&15 within 16x16 tile
#pragma unroll
    for (int mi = 0; mi < 2; ++mi) {
        int rbase = tileM + waveM * 32 + mi * 16 + kgrp * 8;
        float nar[8];
#pragma unroll
        for (int r = 0; r < 8; ++r) nar[r] = na[rbase + r];
#pragma unroll
        for (int ni = 0; ni < 4; ++ni) {
            int col = tileN + waveN * 64 + ni * 16 + laneRow;
            float nbc = nb[col];
#pragma unroll
            for (int r = 0; r < 8; ++r) {
                float denom = fmaxf(nar[r] * nbc, EPSV);
                cosOut[(size_t)(rbase + r) * BN + col] = acc[mi][ni][r] / denom;
            }
        }
    }
}

// ---------------- kernel 3: per-row log-softmax + masked mean ----------------
__global__ __launch_bounds__(256)
void softmax_rows_kernel(const float* __restrict__ cosM, const int* __restrict__ labels,
                         float* __restrict__ per_row) {
    __shared__ float sm[256], ss[256], sps[256], spc[256];
    const int i   = blockIdx.x;
    const int tid = threadIdx.x;
    const int li  = labels[i];
    const float* row = cosM + (size_t)i * BN;

    float m = -INFINITY, s = 0.f, ps = 0.f, pc = 0.f;
    for (int j = tid; j < BN; j += 256) {
        float v  = row[j];
        float nm = fmaxf(m, v);
        s = s * __expf(m - nm) + __expf(v - nm);
        m = nm;
        if (labels[j] == li) { ps += v; pc += 1.f; }
    }
    sm[tid] = m; ss[tid] = s; sps[tid] = ps; spc[tid] = pc;
    __syncthreads();
    for (int off = 128; off > 0; off >>= 1) {
        if (tid < off) {
            float m2 = sm[tid + off], s2 = ss[tid + off];
            float nm = fmaxf(sm[tid], m2);
            ss[tid]  = ss[tid] * __expf(sm[tid] - nm) + s2 * __expf(m2 - nm);
            sm[tid]  = nm;
            sps[tid] += sps[tid + off];
            spc[tid] += spc[tid + off];
        }
        __syncthreads();
    }
    if (tid == 0) {
        // per_row = (m + log(sum exp(x-m))) - sum_pos(x)/count_pos
        per_row[i] = sm[0] + logf(ss[0]) - sps[0] / spc[0];
    }
}

// ---------------- kernel 4: mean over rows -> loss (deterministic) ----------------
__global__ __launch_bounds__(1024)
void finalize_kernel(const float* __restrict__ per_row, float* __restrict__ out) {
    __shared__ float red[1024];
    const int tid = threadIdx.x;
    float s = 0.f;
#pragma unroll
    for (int i = 0; i < BN / 1024; ++i) s += per_row[tid + i * 1024];
    red[tid] = s;
    __syncthreads();
    for (int off = 512; off > 0; off >>= 1) {
        if (tid < off) red[tid] += red[tid + off];
        __syncthreads();
    }
    if (tid == 0) out[0] = red[0] / (float)BN;
}

// ---------------- launcher ----------------
extern "C" void kernel_launch(void* const* d_in, const int* in_sizes, int n_in,
                              void* d_out, int out_size, void* d_ws, size_t ws_size,
                              hipStream_t stream) {
    (void)in_sizes; (void)n_in; (void)out_size; (void)ws_size;
    const int*   labels = (const int*)d_in[0];
    const float* A      = (const float*)d_in[1];
    const float* Bf     = (const float*)d_in[2];

    float* out    = (float*)d_out;
    float* cosOut = out + 1;           // d_out = [loss, cos_score(8192x8192)]

    // workspace layout:
    //   na[8192] | nb[8192] | per_row[8192] (floats)
    //   then bf16 planes: Ahi | Alo | Bhi | Blo, each BN*DK (16 MB) -> 64 MB total
    float* ws      = (float*)d_ws;
    float* na      = ws;
    float* nb      = ws + BN;
    float* per_row = ws + 2 * BN;
    unsigned short* planes = (unsigned short*)(ws + 3 * BN);
    const size_t PLANE = (size_t)BN * DK;
    unsigned short* Ahi = planes;
    unsigned short* Alo = planes + PLANE;
    unsigned short* Bhi = planes + 2 * PLANE;
    unsigned short* Blo = planes + 3 * PLANE;

    preprocess_kernel<<<2 * BN, 256, 0, stream>>>(A, Bf, Ahi, Alo, Bhi, Blo, na, nb);

    dim3 grid(BN / 128, BN / 128);     // 64 x 64 tiles
    gemm_cos_kernel<<<grid, 256, 0, stream>>>(Ahi, Alo, Bhi, Blo, na, nb, cosOut);

    softmax_rows_kernel<<<BN, 256, 0, stream>>>(cosOut, labels, per_row);

    finalize_kernel<<<1, 1024, 0, stream>>>(per_row, out);
}
